// TreeModel_90039694393383
// MI455X (gfx1250) — compile-verified, matrix operand order checked by reading, not run on hardware
//
#include <hip/hip_runtime.h>
#include <stdint.h>

typedef __attribute__((ext_vector_type(16))) __bf16 v16bf;
typedef __attribute__((ext_vector_type(8)))  __bf16 v8bf;
typedef __attribute__((ext_vector_type(8)))  float  v8f;

// ---------------- workspace layout (bytes) ----------------
#define OFF_XBF     0u           // 8192*256 bf16
#define OFF_EMBA    4194304u     // 8192*256 bf16
#define OFF_EMBB    8388608u     // 8192*256 bf16
#define OFF_ENCWT   12582912u    // 4*256*256 bf16   [layer][out][in]
#define OFF_W1T     13107200u    // 31*64*256 bf16   [n][h_out(pad64)][e]
#define OFF_WWT     14123008u    // 31*256*64 bf16   [n][d][h(pad64)]
#define OFF_LFW1T   15138816u    // 32*256*256 bf16  [l][out][in]
#define OFF_WBP     19333120u    // 31*64 f32 (padded)
#define OFF_WBETAP  19341056u    // 31*64 f32 (padded)
#define OFF_PR      19348992u    // 31*8192 f32
#define OFF_YLEAF   20364800u    // 32*8192 f32

union FragU { v16bf f; v8bf h[2]; };

// Load a 16x32 bf16 A-style fragment (also used for B with [N][K] storage).
// ISA 7.12.2: lane<16 holds M=lane, K in {k0..k0+7, k0+16..k0+23};
// lane>=16 holds M=lane-16, K in {k0+8..k0+15, k0+24..k0+31}.
__device__ __forceinline__ v16bf load_frag(const __bf16* base, int ld, int row, int k0, int lane) {
  int m  = lane & 15;
  int hh = (lane >> 4) & 1;
  const __bf16* p = base + (size_t)(row + m) * ld + k0 + hh * 8;
  FragU u;
  u.h[0] = *(const v8bf*)(p);
  u.h[1] = *(const v8bf*)(p + 16);
  return u.f;
}

__device__ __forceinline__ v8f wmma_bf16(v16bf a, v16bf b, v8f c) {
  return __builtin_amdgcn_wmma_f32_16x16x32_bf16(false, a, false, b, (short)0, c, false, false);
}

// ---------------- prep / conversion kernels ----------------
__global__ void cvt_bf16(const float* __restrict__ src, __bf16* __restrict__ dst, int n) {
  int i = blockIdx.x * blockDim.x + threadIdx.x;
  if (i < n) dst[i] = (__bf16)src[i];
}

// dst[b][o][r] = src[b][r][o]  (batched transpose + cvt); R=rows_in, C=cols_in
__global__ void transpose_cvt(const float* __restrict__ src, __bf16* __restrict__ dst,
                              int batch, int R, int C) {
  int i = blockIdx.x * blockDim.x + threadIdx.x;
  int total = batch * R * C;
  if (i >= total) return;
  int b = i / (R * C);
  int rem = i - b * (R * C);
  int o = rem / R;
  int r = rem - o * R;
  dst[i] = (__bf16)src[(size_t)b * R * C + (size_t)r * C + o];
}

// in_W1 [n][256][50] -> W1t [n][64][256], zero-padded rows 50..63
__global__ void prep_W1t(const float* __restrict__ src, __bf16* __restrict__ dst) {
  int i = blockIdx.x * blockDim.x + threadIdx.x;
  if (i >= 31 * 64 * 256) return;
  int n = i / (64 * 256);
  int rem = i - n * (64 * 256);
  int o = rem >> 8;       // h index 0..63
  int k = rem & 255;      // e index
  float v = (o < 50) ? src[(size_t)n * 256 * 50 + (size_t)k * 50 + o] : 0.f;
  dst[i] = (__bf16)v;
}

// in_Ww [n][50][256] -> Wwt [n][256][64], zero-padded k 50..63
__global__ void prep_Wwt(const float* __restrict__ src, __bf16* __restrict__ dst) {
  int i = blockIdx.x * blockDim.x + threadIdx.x;
  if (i >= 31 * 256 * 64) return;
  int n = i / (256 * 64);
  int rem = i - n * (256 * 64);
  int d = rem >> 6;
  int k = rem & 63;
  float v = (k < 50) ? src[(size_t)n * 50 * 256 + (size_t)k * 256 + d] : 0.f;
  dst[i] = (__bf16)v;
}

__global__ void prep_vec(const float* __restrict__ Wb, const float* __restrict__ Wbeta,
                         float* __restrict__ dWb, float* __restrict__ dWbeta) {
  int i = blockIdx.x * blockDim.x + threadIdx.x;
  if (i >= 31 * 64) return;
  int n = i >> 6, c = i & 63;
  dWb[i]    = (c < 50) ? Wb[n * 50 + c]    : 0.f;
  dWbeta[i] = (c < 50) ? Wbeta[n * 50 + c] : 0.f;
}

// ---------------- encoder GEMM: C = relu(A @ Wt^T + bias), all 8192x256 ----------------
__global__ void __launch_bounds__(256)
gemm_bias_relu(const __bf16* __restrict__ A, const __bf16* __restrict__ Wt,
               const float* __restrict__ bias, __bf16* __restrict__ C) {
  int lane = threadIdx.x & 31;
  int wave = threadIdx.x >> 5;
  int row0 = blockIdx.x * 128 + wave * 16;
  int m = lane & 15, hh = lane >> 4;
  for (int t = 0; t < 16; ++t) {
    v8f acc = {};
#pragma unroll
    for (int k0 = 0; k0 < 256; k0 += 32) {
      v16bf aF = load_frag(A, 256, row0, k0, lane);
      v16bf bF = load_frag(Wt, 256, t * 16, k0, lane);
      acc = wmma_bf16(aF, bF, acc);
    }
    int col = t * 16 + m;
    float bv = bias[col];
#pragma unroll
    for (int r = 0; r < 8; ++r) {
      float v = acc[r] + bv;
      v = v > 0.f ? v : 0.f;
      C[(size_t)(row0 + r + 8 * hh) * 256 + col] = (__bf16)v;
    }
  }
}

// ---------------- inner nodes: h -> online softmax(h@Ww+bw) dot x -> sigmoid ----------------
__global__ void __launch_bounds__(256)
inner_nodes(const __bf16* __restrict__ emb, const __bf16* __restrict__ W1t,
            const float* __restrict__ b1, const __bf16* __restrict__ Wwt,
            const float* __restrict__ bw, const float* __restrict__ Wb,
            const float* __restrict__ bb, const float* __restrict__ Wbeta,
            const float* __restrict__ bbeta, const float* __restrict__ x,
            float* __restrict__ pR) {
  __shared__ __bf16 h_lds[128 * 64];
  __shared__ float  sz_lds[128];
  int n = blockIdx.y;
  int rb = blockIdx.x * 128;
  int lane = threadIdx.x & 31;
  int wave = threadIdx.x >> 5;
  int m = lane & 15, hh = lane >> 4;
  int rloc0 = wave * 16;

  // Stage 1: h = relu(emb @ W1t[n]^T + b1)  -> LDS (bf16, 128x64)
  const __bf16* W1n = W1t + (size_t)n * 64 * 256;
#pragma unroll
  for (int t = 0; t < 4; ++t) {
    v8f acc = {};
#pragma unroll
    for (int k0 = 0; k0 < 256; k0 += 32) {
      v16bf aF = load_frag(emb, 256, rb + rloc0, k0, lane);
      v16bf bF = load_frag(W1n, 256, t * 16, k0, lane);
      acc = wmma_bf16(aF, bF, acc);
    }
    int col = t * 16 + m;
    float bv = (col < 50) ? b1[n * 50 + col] : 0.f;
#pragma unroll
    for (int r = 0; r < 8; ++r) {
      float v = acc[r] + bv;
      v = v > 0.f ? v : 0.f;
      h_lds[(rloc0 + r + 8 * hh) * 64 + col] = (__bf16)v;
    }
  }
  __syncthreads();

  // Stage 2: logits tiles via WMMA (K=64); online softmax + dot with x, in registers.
  const __bf16* Wwn = Wwt + (size_t)n * 256 * 64;
  v16bf aF0 = load_frag(h_lds, 64, rloc0, 0, lane);
  v16bf aF1 = load_frag(h_lds, 64, rloc0, 32, lane);
  float mrun[8], Zrun[8], Srun[8];
#pragma unroll
  for (int r = 0; r < 8; ++r) { mrun[r] = -3.4e38f; Zrun[r] = 0.f; Srun[r] = 0.f; }
  for (int t = 0; t < 16; ++t) {
    v8f acc = {};
    acc = wmma_bf16(aF0, load_frag(Wwn, 64, t * 16, 0, lane), acc);
    acc = wmma_bf16(aF1, load_frag(Wwn, 64, t * 16, 32, lane), acc);
    int col = t * 16 + m;
    float bwv = bw[n * 256 + col];
#pragma unroll
    for (int r = 0; r < 8; ++r) {
      int row = rb + rloc0 + r + 8 * hh;
      float v = acc[r] + bwv;
      float tm = v;                       // 16-lane max (masks<16 keep halves separate)
#pragma unroll
      for (int o = 1; o < 16; o <<= 1) tm = fmaxf(tm, __shfl_xor(tm, o, 32));
      float mn = fmaxf(mrun[r], tm);
      float sc = __expf(mrun[r] - mn);
      float e  = __expf(v - mn);
      float xv = x[(size_t)row * 256 + col];
      float se = e, sx = e * xv;
#pragma unroll
      for (int o = 1; o < 16; o <<= 1) { se += __shfl_xor(se, o, 32); sx += __shfl_xor(sx, o, 32); }
      Zrun[r] = Zrun[r] * sc + se;
      Srun[r] = Srun[r] * sc + sx;
      mrun[r] = mn;
    }
  }
  if (m == 0) {
#pragma unroll
    for (int r = 0; r < 8; ++r)
      sz_lds[rloc0 + r + 8 * hh] = Srun[r] / Zrun[r];
  }
  __syncthreads();

  // Stage 3: b/beta dots + sigmoid, one thread per row
  if (threadIdx.x < 128) {
    int r = threadIdx.x;
    float bv = bb[n], betav = bbeta[n];
#pragma unroll 8
    for (int c = 0; c < 64; ++c) {
      float hv = (float)h_lds[r * 64 + c];
      bv    += hv * Wb[n * 64 + c];
      betav += hv * Wbeta[n * 64 + c];
    }
    float arg = betav * (sz_lds[r] + bv);
    pR[(size_t)n * 8192 + rb + r] = 1.f / (1.f + __expf(-arg));
  }
}

// ---------------- leaf MLPs: y = relu(emb@W1^T + b1) . W2 + b2 ----------------
__global__ void __launch_bounds__(256)
leaf_nodes(const __bf16* __restrict__ emb, const __bf16* __restrict__ Wt,
           const float* __restrict__ b1, const float* __restrict__ W2,
           const float* __restrict__ b2, float* __restrict__ y_leaf) {
  int l = blockIdx.y;
  int rb = blockIdx.x * 128;
  int lane = threadIdx.x & 31;
  int wave = threadIdx.x >> 5;
  int m = lane & 15, hh = lane >> 4;
  int row0 = rb + wave * 16;
  const __bf16* Wl = Wt + (size_t)l * 256 * 256;
  float ypart[8];
#pragma unroll
  for (int r = 0; r < 8; ++r) ypart[r] = 0.f;
  for (int t = 0; t < 16; ++t) {
    v8f acc = {};
#pragma unroll
    for (int k0 = 0; k0 < 256; k0 += 32) {
      v16bf aF = load_frag(emb, 256, row0, k0, lane);
      v16bf bF = load_frag(Wl, 256, t * 16, k0, lane);
      acc = wmma_bf16(aF, bF, acc);
    }
    int col = t * 16 + m;
    float bv = b1[l * 256 + col];
    float w2 = W2[l * 256 + col];
#pragma unroll
    for (int r = 0; r < 8; ++r) {
      float v = acc[r] + bv;
      v = v > 0.f ? v : 0.f;
      ypart[r] += v * w2;
    }
  }
#pragma unroll
  for (int r = 0; r < 8; ++r) {
    float s = ypart[r];
#pragma unroll
    for (int o = 1; o < 16; o <<= 1) s += __shfl_xor(s, o, 32);
    if (m == 0)
      y_leaf[(size_t)l * 8192 + row0 + r + 8 * hh] = s + b2[l];
  }
}

// ---------------- tree combine ----------------
__global__ void combine(const float* __restrict__ pR, const float* __restrict__ y_leaf,
                        float* __restrict__ out) {
  int b = blockIdx.x * blockDim.x + threadIdx.x;
  if (b >= 8192) return;
  float prob[32];
  prob[0] = 1.f;
  int idx = 0;
#pragma unroll
  for (int level = 0; level < 5; ++level) {
    int cnt = 1 << level;
#pragma unroll 16
    for (int i = cnt - 1; i >= 0; --i) {
      float p = pR[(size_t)(idx + i) * 8192 + b];
      float base = prob[i];
      prob[2 * i]     = base * (1.f - p);
      prob[2 * i + 1] = base * p;
    }
    idx += cnt;
  }
  float acc = 0.f;
#pragma unroll
  for (int l = 0; l < 32; ++l) acc += prob[l] * y_leaf[(size_t)l * 8192 + b];
  out[b] = acc;
}

// ---------------- launcher ----------------
extern "C" void kernel_launch(void* const* d_in, const int* in_sizes, int n_in,
                              void* d_out, int out_size, void* d_ws, size_t ws_size,
                              hipStream_t stream) {
  const float* x        = (const float*)d_in[0];
  const float* enc_W    = (const float*)d_in[1];
  const float* enc_b    = (const float*)d_in[2];
  const float* in_W1    = (const float*)d_in[3];
  const float* in_b1    = (const float*)d_in[4];
  const float* in_Ww    = (const float*)d_in[5];
  const float* in_bw    = (const float*)d_in[6];
  const float* in_Wb    = (const float*)d_in[7];
  const float* in_bb    = (const float*)d_in[8];
  const float* in_Wbeta = (const float*)d_in[9];
  const float* in_bbeta = (const float*)d_in[10];
  const float* lf_W1    = (const float*)d_in[11];
  const float* lf_b1    = (const float*)d_in[12];
  const float* lf_W2    = (const float*)d_in[13];
  const float* lf_b2    = (const float*)d_in[14];
  float* out = (float*)d_out;

  char* ws = (char*)d_ws;
  __bf16* x_bf    = (__bf16*)(ws + OFF_XBF);
  __bf16* embA    = (__bf16*)(ws + OFF_EMBA);
  __bf16* embB    = (__bf16*)(ws + OFF_EMBB);
  __bf16* encWt   = (__bf16*)(ws + OFF_ENCWT);
  __bf16* W1t     = (__bf16*)(ws + OFF_W1T);
  __bf16* Wwt     = (__bf16*)(ws + OFF_WWT);
  __bf16* lfW1t   = (__bf16*)(ws + OFF_LFW1T);
  float*  Wb_p    = (float*)(ws + OFF_WBP);
  float*  Wbeta_p = (float*)(ws + OFF_WBETAP);
  float*  pR      = (float*)(ws + OFF_PR);
  float*  y_leaf  = (float*)(ws + OFF_YLEAF);

  // prep
  cvt_bf16<<<(8192 * 256 + 255) / 256, 256, 0, stream>>>(x, x_bf, 8192 * 256);
  transpose_cvt<<<(4 * 65536 + 255) / 256, 256, 0, stream>>>(enc_W, encWt, 4, 256, 256);
  prep_W1t<<<(31 * 64 * 256 + 255) / 256, 256, 0, stream>>>(in_W1, W1t);
  prep_Wwt<<<(31 * 256 * 64 + 255) / 256, 256, 0, stream>>>(in_Ww, Wwt);
  transpose_cvt<<<(32 * 65536 + 255) / 256, 256, 0, stream>>>(lf_W1, lfW1t, 32, 256, 256);
  prep_vec<<<(31 * 64 + 255) / 256, 256, 0, stream>>>(in_Wb, in_Wbeta, Wb_p, Wbeta_p);

  // encoder: 4 relu dense layers (ping-pong)
  gemm_bias_relu<<<64, 256, 0, stream>>>(x_bf, encWt + 0 * 65536, enc_b + 0,   embA);
  gemm_bias_relu<<<64, 256, 0, stream>>>(embA, encWt + 1 * 65536, enc_b + 256, embB);
  gemm_bias_relu<<<64, 256, 0, stream>>>(embB, encWt + 2 * 65536, enc_b + 512, embA);
  gemm_bias_relu<<<64, 256, 0, stream>>>(embA, encWt + 3 * 65536, enc_b + 768, embB);

  // inner nodes + leaves (independent; both read embB)
  inner_nodes<<<dim3(64, 31), 256, 0, stream>>>(embB, W1t, in_b1, Wwt, in_bw,
                                                Wb_p, in_bb, Wbeta_p, in_bbeta, x, pR);
  leaf_nodes<<<dim3(64, 32), 256, 0, stream>>>(embB, lfW1t, lf_b1, lf_W2, lf_b2, y_leaf);

  // tree combine
  combine<<<32, 256, 0, stream>>>(pR, y_leaf, out);
}
